// RNN_7121055776970
// MI455X (gfx1250) — matrix operand order repaired and checked by reading, not compiled, not run on hardware
//
#include <hip/hip_runtime.h>
#include <math.h>
#include <stdint.h>

#define IN_DIM 128
#define H_DIM  256
#define T_LEN  1024
#define BS     128
#define HP     (H_DIM + 8)    // bf16 LDS row stride, phase-2 WhT / h
#define XBP    (IN_DIM + 8)   // bf16 LDS row stride, phase-1 x tile (272B, 16B-aligned)
#define WXP    (IN_DIM + 8)   // bf16 LDS row stride, phase-1 WxT

typedef __attribute__((ext_vector_type(16))) __bf16 v16bf;
typedef __attribute__((ext_vector_type(8)))  __bf16 v8bf;
typedef __attribute__((ext_vector_type(8)))  float  v8f;

// Async DMA: 16 bytes global -> LDS, tracked by ASYNCcnt (gfx1250).
// lds_off = wave-relative LDS byte address (low 32 bits of the flat
// shared-aperture address; ISA 10.2).
__device__ __forceinline__ void async_copy16(uint32_t lds_off, const void* gaddr) {
    asm volatile("global_load_async_to_lds_b128 %0, %1, off"
                 :: "v"(lds_off), "v"(gaddr) : "memory");
}

__device__ __forceinline__ void wait_async0() {
#if __has_builtin(__builtin_amdgcn_s_wait_asynccnt)
    __builtin_amdgcn_s_wait_asynccnt(0);
#else
    asm volatile("s_wait_asynccnt 0" ::: "memory");
#endif
}

// Build a 16x32 WMMA operand register set from a padded bf16 LDS row/col:
// two contiguous 16B ds_load_b128 + shuffle. lo holds elems {base..base+7},
// hi holds {base+ofs..}, matching the ISA 7.12.2 bf16 layouts.
__device__ __forceinline__ v16bf ld_pair(const __bf16* p, int ofs) {
    v8bf lo = *(const v8bf*)p;
    v8bf hi = *(const v8bf*)(p + ofs);
    return __builtin_shufflevector(lo, hi,
        0,1,2,3,4,5,6,7,8,9,10,11,12,13,14,15);
}

// ---------------------------------------------------------------------------
// Phase 1: pre = x @ Wx + (bx + bh) -> d_out (f32).
// Block = 128 M-rows x 256 N-cols, 1024 blocks, 512 threads (16 waves).
// x tile (64KB f32) DMA'd into an LDS staging region by the async engine,
// converted once to padded bf16; Wx staged transposed bf16. Wave computes
// 2x4 = 8 16x16 tiles (A reused across N, B across M): 32 WMMAs/wave, all
// operands via contiguous ds_load_b128 -> low register pressure, no spills.
// ---------------------------------------------------------------------------
__global__ __launch_bounds__(512) void pre_gemm_kernel(
    const float* __restrict__ x, const float* __restrict__ Wx,
    const float* __restrict__ bx, const float* __restrict__ bh,
    float* __restrict__ out)
{
    __shared__ float  sXf[128 * IN_DIM];     // 65,536 B (async DMA stage)
    __shared__ __bf16 sXb[128 * XBP];        // 34,816 B (padded bf16 x)
    __shared__ __bf16 sWxT[H_DIM * WXP];     // 69,632 B (padded bf16 Wx^T)

    const int tid  = threadIdx.x;
    const int lane = tid & 31;
    const int wave = tid >> 5;

    // 1) Async DMA the 128x128 f32 x tile (contiguous 64KB).
    {
        const float* xblk = x + (size_t)blockIdx.x * 128 * IN_DIM;
        const uint32_t ldsb = (uint32_t)(uintptr_t)&sXf[0];
        #pragma unroll
        for (int j = 0; j < 8; ++j) {        // 4096 16B chunks / 512 threads
            const int idx = tid + j * 512;
            async_copy16(ldsb + idx * 16, xblk + idx * 4);
        }
    }
    // 2) Stage Wx transposed (f32 -> bf16), coalesced global reads
    //    (overlaps the async DMA).
    for (int i = tid; i < IN_DIM * H_DIM; i += 512) {
        const int n = i & (H_DIM - 1);
        const int k = i >> 8;
        sWxT[n * WXP + k] = (__bf16)Wx[(size_t)k * H_DIM + n];
    }
    wait_async0();
    __syncthreads();

    // 3) One conversion sweep: f32 stage -> padded bf16 A-tile.
    for (int i = tid; i < 128 * IN_DIM; i += 512) {
        const int r = i >> 7;
        const int k = i & (IN_DIM - 1);
        sXb[r * XBP + k] = (__bf16)sXf[i];
    }
    __syncthreads();

    const int row = lane & 15;
    const int hi  = lane >> 4;
    const int wm  = wave >> 2;               // 4 m-groups of 2 tiles (32 rows)
    const int wn  = wave & 3;                // 4 n-groups of 4 tiles (64 cols)

    v8f acc[2][4] = {};
    #pragma unroll
    for (int c = 0; c < IN_DIM / 32; ++c) {
        const int abase = c * 32 + hi * 8;
        const int bbase = c * 32 + hi * 8;
        v16bf a[2], b[4];
        #pragma unroll
        for (int mi = 0; mi < 2; ++mi)
            a[mi] = ld_pair(sXb + (wm * 32 + mi * 16 + row) * XBP + abase, 16);
        #pragma unroll
        for (int ni = 0; ni < 4; ++ni)
            b[ni] = ld_pair(sWxT + (wn * 64 + ni * 16 + row) * WXP + bbase, 16);
        #pragma unroll
        for (int mi = 0; mi < 2; ++mi)
            #pragma unroll
            for (int ni = 0; ni < 4; ++ni)
                acc[mi][ni] = __builtin_amdgcn_wmma_f32_16x16x32_bf16(
                    false, a[mi], false, b[ni], (short)0, acc[mi][ni], false, false);
    }

    float* oblk = out + (size_t)blockIdx.x * 128 * H_DIM;
    #pragma unroll
    for (int ni = 0; ni < 4; ++ni) {
        const int col = wn * 64 + ni * 16 + row;
        const float bias = bx[col] + bh[col];
        #pragma unroll
        for (int mi = 0; mi < 2; ++mi) {
            float* orow = oblk + (size_t)(wm * 32 + mi * 16) * H_DIM + col;
            #pragma unroll
            for (int v = 0; v < 8; ++v)
                orow[(size_t)(v + hi * 8) * H_DIM] = acc[mi][ni][v] + bias;
        }
    }
}

// ---------------------------------------------------------------------------
// Phase 2: sequential recurrence. 8 blocks; block owns batch rows b0..b0+15
// (independent chains -> no inter-block sync). 16 waves, one 16-col N-tile
// each. Wh transposed+padded in LDS (bf16), h double-buffered in LDS, and
// the pre[t] tile (16KB, contiguous) DMA'd in with the async engine so the
// fetch of pre[t+1] overlaps step t's 8-deep WMMA chain.
// ---------------------------------------------------------------------------
__global__ __launch_bounds__(512) void rnn_rec_kernel(
    const float* __restrict__ h0, const float* __restrict__ Wh,
    float* __restrict__ out)
{
    __shared__ __bf16 sWhT[H_DIM * HP];
    __shared__ __bf16 sh[2][16 * HP];
    __shared__ float  sPre[2][16 * H_DIM];

    const int tid  = threadIdx.x;
    const int lane = tid & 31;
    const int wave = tid >> 5;
    const int b0   = blockIdx.x * 16;

    {
        const float* gsrc = out + (size_t)b0 * H_DIM;   // t=0 tile
        const uint32_t ldsb = (uint32_t)(uintptr_t)&sPre[0][0];
        #pragma unroll
        for (int j = 0; j < 2; ++j) {
            const int idx = tid + j * 512;
            async_copy16(ldsb + idx * 16, gsrc + idx * 4);
        }
    }
    for (int i = tid; i < H_DIM * H_DIM; i += 512) {
        const int n = i & (H_DIM - 1);
        const int k = i >> 8;
        sWhT[n * HP + k] = (__bf16)Wh[(size_t)k * H_DIM + n];
    }
    for (int i = tid; i < 16 * H_DIM; i += 512) {
        const int r = i >> 8;
        const int c = i & (H_DIM - 1);
        sh[0][r * HP + c] = (__bf16)h0[(size_t)(b0 + r) * H_DIM + c];
    }
    wait_async0();
    __syncthreads();

    const int row   = lane & 15;
    const int hi    = lane >> 4;
    const int col   = wave * 16 + row;
    const __bf16* bcol = sWhT + (size_t)col * HP;

    int p = 0;
    for (int t = 0; t < T_LEN; ++t) {
        if (t + 1 < T_LEN) {
            const float* gsrc = out + ((size_t)(t + 1) * BS + b0) * H_DIM;
            const uint32_t ldsb = (uint32_t)(uintptr_t)&sPre[(t + 1) & 1][0];
            #pragma unroll
            for (int j = 0; j < 2; ++j) {
                const int idx = tid + j * 512;
                async_copy16(ldsb + idx * 16, gsrc + idx * 4);
            }
        }

        const __bf16* arow = sh[p] + row * HP;
        v8f acc = {};
        #pragma unroll
        for (int c = 0; c < H_DIM / 32; ++c) {
            v16bf a = ld_pair(arow + c * 32 + hi * 8, 16);
            v16bf b = ld_pair(bcol + c * 32 + hi * 8, 16);
            acc = __builtin_amdgcn_wmma_f32_16x16x32_bf16(
                false, a, false, b, (short)0, acc, false, false);
        }

        const float* pbuf = &sPre[t & 1][0];
        float* obase = out + ((size_t)t * BS + b0) * H_DIM + col;
        __bf16* hnew = sh[1 ^ p];
        #pragma unroll
        for (int v = 0; v < 8; ++v) {
            const int m   = v + hi * 8;
            const float z = tanhf(pbuf[m * H_DIM + col] + acc[v]);
            obase[(size_t)m * H_DIM] = z;
            hnew[m * HP + col] = (__bf16)z;
        }
        if (t != 0) p ^= 1;   // quirk: step 1 re-reads the initial h
        wait_async0();
        __syncthreads();
    }
}

extern "C" void kernel_launch(void* const* d_in, const int* in_sizes, int n_in,
                              void* d_out, int out_size, void* d_ws, size_t ws_size,
                              hipStream_t stream) {
    (void)in_sizes; (void)n_in; (void)out_size; (void)d_ws; (void)ws_size;
    const float* x  = (const float*)d_in[0];
    const float* h  = (const float*)d_in[1];
    const float* Wx = (const float*)d_in[2];
    const float* bx = (const float*)d_in[3];
    const float* Wh = (const float*)d_in[4];
    const float* bh = (const float*)d_in[5];
    float* out = (float*)d_out;

    pre_gemm_kernel<<<1024, 512, 0, stream>>>(x, Wx, bx, bh, out);
    rnn_rec_kernel<<<8, 512, 0, stream>>>(h, Wh, out);
}
